// MambaBlock_54254026883130
// MI455X (gfx1250) — matrix operand order repaired and checked by reading, not compiled, not run on hardware
//
#include <hip/hip_runtime.h>
#include <hip/hip_bf16.h>
#include <stdint.h>

// ---------------------------------------------------------------------------
// MambaBlock on gfx1250 (MI455X): bf16 WMMA GEMMs (LDS-free, L2-resident
// pre-transposed weights, software-pipelined) + fp32 selective scan
// ---------------------------------------------------------------------------

typedef __attribute__((ext_vector_type(16))) __bf16 v16bf;
typedef __attribute__((ext_vector_type(8)))  float  v8f;

#define DEV static __device__ __forceinline__

namespace {
constexpr int B_   = 2;
constexpr int L_   = 1024;
constexpr int LP_  = L_ + 3;    // causal-padded rows per batch
constexpr int DM_  = 2048;
constexpr int DIN_ = 2048;
constexpr int NS_  = 16;    // state dim N
constexpr int DTR_ = 128;
constexpr int KC_  = 4;     // conv taps
constexpr int M_   = B_ * L_;   // 2048 GEMM rows everywhere

// 256 threads = 8 waves (2x4); block tile 128(M) x 128(N); wave tile 64x32.
constexpr int BM = 128;
constexpr int BN = 128;
constexpr int BK = 32;
}

DEV unsigned short f32_to_bf16(float f) {
  unsigned int u = __float_as_uint(f);
  unsigned int r = (u + 0x7FFFu + ((u >> 16) & 1u)) >> 16;  // RNE
  return (unsigned short)r;
}
DEV float sigmoidf_(float x) { return 1.0f / (1.0f + __expf(-x)); }
DEV float siluf_(float x)    { return x * sigmoidf_(x); }
DEV float softplusf_(float x){ return (x > 20.0f) ? x : log1pf(__expf(x)); }

union Frag {
  v16bf v;
  uint4 q[2];
  unsigned short s[16];
};

// A fragment (16x32): base = A + row*lda + g*8 ; elems 0..7 -> K+g*8+j,
// elems 8..15 -> K+16+g*8+j  (two b128 loads, 32B apart)
DEV v16bf load_a_ptr(const unsigned short* base, int k0) {
  const uint4* p = reinterpret_cast<const uint4*>(base + k0);
  Frag f;
  f.q[0] = p[0];
  f.q[1] = p[2];
  return f.v;
}
// B fragment (32x16) from BT[n][k]: base = BT + col*Kk + g*8 ; elems 0..7 -> K+g*8+j,
// elems 8..15 -> K+16+g*8+j
DEV v16bf load_b_ptr(const unsigned short* base, int k0) {
  const uint4* p = reinterpret_cast<const uint4*>(base + k0);
  Frag f;
  f.q[0] = p[0];
  f.q[1] = p[2];
  return f.v;
}

DEV void load_frags(v16bf (&a)[4], v16bf (&b)[2],
                    const unsigned short* const (&abase)[4],
                    const unsigned short* const (&bbase)[2], int k0) {
#pragma unroll
  for (int tm = 0; tm < 4; ++tm) a[tm] = load_a_ptr(abase[tm], k0);
#pragma unroll
  for (int tn = 0; tn < 2; ++tn) b[tn] = load_b_ptr(bbase[tn], k0);
}

DEV void mma_step(v8f (&acc)[4][2], const v16bf (&a)[4], const v16bf (&b)[2]) {
#pragma unroll
  for (int tm = 0; tm < 4; ++tm)
#pragma unroll
    for (int tn = 0; tn < 2; ++tn)
      acc[tm][tn] = __builtin_amdgcn_wmma_f32_16x16x32_bf16(
          false, a[tm], false, b[tn], (short)0, acc[tm][tn], false, false);
}

enum { EPI_SPLIT_XZ = 0, EPI_DBC = 1, EPI_SOFTPLUS = 2, EPI_ADD = 3, EPI_CONV = 4 };

// One GEMM template for all 5 matmuls. Requires: Kk % 64 == 0, Kk >= 128,
// M % 128 == 0. Partial N handled by column clamp + epilogue guard.
template <int EPI>
__global__ __launch_bounds__(256) void gemm_bf16_kernel(
    const unsigned short* __restrict__ A,   // row-major bf16 (conv: padded)
    const unsigned short* __restrict__ BT,  // Nn x Kk bf16 (transposed weight)
    int Nn, int Kk, int lda,
    const float* __restrict__ aux,          // bias (per col) or residual (per elem)
    float* __restrict__ out_f,
    float* __restrict__ out_f2,
    unsigned short* __restrict__ out_bf) {
  const int tid  = threadIdx.x;
  const int lane = tid & 31;
  const int g    = lane >> 4;
  const int w    = tid >> 5;
  const int wm   = w >> 2;       // 0..1
  const int wn   = w & 3;        // 0..3
  const int blockM = blockIdx.y * BM;
  const int blockN = blockIdx.x * BN;

  // per-tile base pointers (computed once; loop address math is +k0 only)
  const unsigned short* abase[4];
#pragma unroll
  for (int tm = 0; tm < 4; ++tm) {
    int m = blockM + wm * 64 + tm * 16 + (lane & 15);
    // conv reads the causal-padded activation: row = b*(L+3) + l  (== l-3 shift)
    int row = (EPI == EPI_CONV) ? ((m >> 10) * LP_ + (m & (L_ - 1))) : m;
    abase[tm] = A + (size_t)row * lda + g * 8;
  }
  const unsigned short* bbase[2];
#pragma unroll
  for (int tn = 0; tn < 2; ++tn) {
    int c = blockN + wn * 32 + tn * 16 + (lane & 15);
    if (c >= Nn) c = Nn - 1;                 // clamp (epilogue guards writes)
    bbase[tn] = BT + (size_t)c * Kk + g * 8;
  }

  v8f acc[4][2] = {};
  v16bf a0[4], b0[2], a1[4], b1[2];
  load_frags(a0, b0, abase, bbase, 0);
  load_frags(a1, b1, abase, bbase, BK);
  int k0 = 0;
  for (; k0 + 2 * BK < Kk; k0 += 2 * BK) {
    mma_step(acc, a0, b0);
    load_frags(a0, b0, abase, bbase, k0 + 2 * BK);
    mma_step(acc, a1, b1);
    load_frags(a1, b1, abase, bbase, k0 + 3 * BK);
  }
  mma_step(acc, a0, b0);
  mma_step(acc, a1, b1);

  // epilogue: VGPR i -> (M = i + 8*(lane>>4), N = lane&15) within 16x16 tile
  const int colo = lane & 15;
  const int rowo = g * 8;
#pragma unroll
  for (int tm = 0; tm < 4; ++tm) {
#pragma unroll
    for (int tn = 0; tn < 2; ++tn) {
      const int bm = blockM + wm * 64 + tm * 16;
      const int bn = blockN + wn * 32 + tn * 16;
#pragma unroll
      for (int i = 0; i < 8; ++i) {
        const int m = bm + rowo + i;
        const int n = bn + colo;
        float v = acc[tm][tn][i];
        if (EPI == EPI_SPLIT_XZ) {
          // cols [0,DIN): xc -> padded bf16 buffer; cols [DIN,2DIN): r (fp32)
          if (n < DIN_) {
            int b = m >> 10, l = m & (L_ - 1);
            out_bf[(size_t)(b * LP_ + 3 + l) * DIN_ + n] = f32_to_bf16(v);
          } else {
            out_f[(size_t)m * DIN_ + (n - DIN_)] = v;
          }
        } else if (EPI == EPI_DBC) {
          if (n >= Nn) continue;
          // cols [0,128): delta-input (bf16); cols [128,160): B,C (fp32)
          if (n < DTR_) out_bf[(size_t)m * DTR_ + n] = f32_to_bf16(v);
          else          out_f [(size_t)m * (2 * NS_) + (n - DTR_)] = v;
        } else if (EPI == EPI_SOFTPLUS) {
          out_f[(size_t)m * DIN_ + n] = softplusf_(v + aux[n]);
        } else if (EPI == EPI_CONV) {
          float xs = v + aux[n];
          size_t o = (size_t)m * DIN_ + n;
          out_f[o] = xs;
          float uu = siluf_(xs);
          out_f2[o] = uu;
          out_bf[o] = f32_to_bf16(uu);
        } else {  // EPI_ADD: final GEMM + xs residual
          out_f[(size_t)m * DM_ + n] = v + aux[(size_t)m * DM_ + n];
        }
      }
    }
  }
}

// selective scan: one thread per (b,d); h[16] in registers, serial over L
__global__ __launch_bounds__(256) void scan_kernel(
    const float* __restrict__ delta_f, const float* __restrict__ u_f,
    const float* __restrict__ bc_f,    const float* __restrict__ r_f,
    const float* __restrict__ A_log,   const float* __restrict__ Dp,
    unsigned short* __restrict__ g_bf) {
  const int idx = blockIdx.x * blockDim.x + threadIdx.x;
  if (idx >= B_ * DIN_) return;
  const int b = idx / DIN_;
  const int d = idx % DIN_;

  float Ad[NS_];
#pragma unroll
  for (int n = 0; n < NS_; ++n) Ad[n] = -__expf(A_log[d * NS_ + n]);
  float h[NS_];
#pragma unroll
  for (int n = 0; n < NS_; ++n) h[n] = 0.0f;
  const float Dd = Dp[d];

  for (int l = 0; l < L_; ++l) {
    const size_t row = (size_t)(b * L_ + l);
    const float dlt = delta_f[row * DIN_ + d];
    const float uu  = u_f[row * DIN_ + d];
    const float du  = dlt * uu;
    float y = 0.0f;
#pragma unroll
    for (int n = 0; n < NS_; ++n) {
      const float Bn = bc_f[row * (2 * NS_) + n];
      const float Cn = bc_f[row * (2 * NS_) + NS_ + n];
      const float dA = __expf(dlt * Ad[n]);
      h[n] = dA * h[n] + du * Bn;
      y += h[n] * Cn;
    }
    const float rv = r_f[row * DIN_ + d];
    const float g  = (y + uu * Dd) * (rv * sigmoidf_(rv));
    g_bf[row * DIN_ + d] = f32_to_bf16(g);
  }
}

__global__ void convert_f32_bf16(const float* __restrict__ s,
                                 unsigned short* __restrict__ d, size_t n) {
  size_t i = (size_t)blockIdx.x * blockDim.x + threadIdx.x;
  size_t stride = (size_t)gridDim.x * blockDim.x;
  for (; i < n; i += stride) d[i] = f32_to_bf16(s[i]);
}

// zero the 3 causal pad rows at the head of each batch in xcpad
__global__ void zero_pad_rows(unsigned short* __restrict__ xcpad) {
  int i = blockIdx.x * blockDim.x + threadIdx.x;
  int per = 3 * DIN_;
  if (i >= B_ * per) return;
  int b = i / per, rem = i % per;
  xcpad[(size_t)b * LP_ * DIN_ + rem] = 0;
}

// transpose-pack: dst[n*K + k] = bf16(src[k*N + n])   (weights -> BT layout)
__global__ void pack_transpose_bf16(const float* __restrict__ src,
                                    unsigned short* __restrict__ dst,
                                    int Nn, int Kk) {
  size_t idx = (size_t)blockIdx.x * blockDim.x + threadIdx.x;
  size_t total = (size_t)Nn * Kk;
  size_t stride = (size_t)gridDim.x * blockDim.x;
  for (; idx < total; idx += stride) {
    int k = (int)(idx % Kk);
    int n = (int)(idx / Kk);
    dst[idx] = f32_to_bf16(src[(size_t)k * Nn + n]);
  }
}

// conv_w (O=DIN, I=DIN, K=4) -> wpackT[o * (KC*DIN) + (k*DIN + i)]
__global__ void pack_convw_kernel(const float* __restrict__ conv_w,
                                  unsigned short* __restrict__ wpackT) {
  size_t idx = (size_t)blockIdx.x * blockDim.x + threadIdx.x;
  size_t total = (size_t)DIN_ * KC_ * DIN_;
  size_t stride = (size_t)gridDim.x * blockDim.x;
  for (; idx < total; idx += stride) {
    int o   = (int)(idx >> 13);            // / (KC_*DIN_)
    int rem = (int)(idx & (KC_ * DIN_ - 1));
    int k   = rem >> 11;                   // / DIN_
    int i   = rem & (DIN_ - 1);
    wpackT[idx] = f32_to_bf16(conv_w[((size_t)o * DIN_ + i) * KC_ + k]);
  }
}

extern "C" void kernel_launch(void* const* d_in, const int* in_sizes, int n_in,
                              void* d_out, int out_size, void* d_ws, size_t ws_size,
                              hipStream_t stream) {
  (void)in_sizes; (void)n_in; (void)out_size; (void)ws_size;
  const float* x      = (const float*)d_in[0];
  const float* W_in   = (const float*)d_in[1];
  const float* conv_w = (const float*)d_in[2];
  const float* conv_b = (const float*)d_in[3];
  const float* W_x    = (const float*)d_in[4];
  const float* W_dt   = (const float*)d_in[5];
  const float* b_dt   = (const float*)d_in[6];
  const float* A_log  = (const float*)d_in[7];
  const float* Dp     = (const float*)d_in[8];
  const float* W_out  = (const float*)d_in[9];
  float* out = (float*)d_out;

  size_t off = 0;
  auto take = [&](size_t bytes) -> void* {
    void* p = (void*)((char*)d_ws + off);
    off += (bytes + 255) & ~(size_t)255;
    return p;
  };
  unsigned short* x_bf     = (unsigned short*)take((size_t)M_ * DM_ * 2);
  unsigned short* WinT_bf  = (unsigned short*)take((size_t)2 * DIN_ * DM_ * 2);
  unsigned short* xcpad    = (unsigned short*)take((size_t)B_ * LP_ * DIN_ * 2);
  float*          r_f      = (float*)take((size_t)M_ * DIN_ * 4);
  unsigned short* wpackT   = (unsigned short*)take((size_t)DIN_ * KC_ * DIN_ * 2);
  float*          xs_f     = (float*)take((size_t)M_ * DIN_ * 4);
  float*          u_f      = (float*)take((size_t)M_ * DIN_ * 4);
  unsigned short* u_bf     = (unsigned short*)take((size_t)M_ * DIN_ * 2);
  unsigned short* WxT_bf   = (unsigned short*)take((size_t)(DTR_ + 2 * NS_) * DIN_ * 2);
  unsigned short* ddt_bf   = (unsigned short*)take((size_t)M_ * DTR_ * 2);
  float*          bc_f     = (float*)take((size_t)M_ * 2 * NS_ * 4);
  unsigned short* WdtT_bf  = (unsigned short*)take((size_t)DIN_ * DTR_ * 2);
  float*          delta_f  = (float*)take((size_t)M_ * DIN_ * 4);
  unsigned short* g_bf     = (unsigned short*)take((size_t)M_ * DIN_ * 2);
  unsigned short* WoutT_bf = (unsigned short*)take((size_t)DM_ * DIN_ * 2);

  // 1) one-time packs: x -> bf16 (row major); weights -> bf16 transposed (BT)
  {
    size_t n = (size_t)M_ * DM_;
    convert_f32_bf16<<<(int)((n + 2047) / 2048), 256, 0, stream>>>(x, x_bf, n);
  }
  pack_transpose_bf16<<<4096, 256, 0, stream>>>(W_in,  WinT_bf,  2 * DIN_, DM_);
  pack_transpose_bf16<<<512,  256, 0, stream>>>(W_x,   WxT_bf,   DTR_ + 2 * NS_, DIN_);
  pack_transpose_bf16<<<512,  256, 0, stream>>>(W_dt,  WdtT_bf,  DIN_, DTR_);
  pack_transpose_bf16<<<4096, 256, 0, stream>>>(W_out, WoutT_bf, DM_, DIN_);
  pack_convw_kernel<<<4096, 256, 0, stream>>>(conv_w, wpackT);
  zero_pad_rows<<<(B_ * 3 * DIN_ + 255) / 256, 256, 0, stream>>>(xcpad);

  const dim3 blk(256);
  // 2) xz = x @ W_in ; split -> xc (bf16, causal-padded), r (fp32)
  gemm_bf16_kernel<EPI_SPLIT_XZ>
      <<<dim3((2 * DIN_) / BN, M_ / BM), blk, 0, stream>>>(
          x_bf, WinT_bf, 2 * DIN_, DM_, DM_, nullptr, r_f, nullptr, xcpad);

  // 3) causal dense conv as contiguous K=8192 GEMM over the padded activation
  gemm_bf16_kernel<EPI_CONV>
      <<<dim3(DIN_ / BN, M_ / BM), blk, 0, stream>>>(
          xcpad, wpackT, DIN_, KC_ * DIN_, DIN_, conv_b, xs_f, u_f, u_bf);

  // 4) dbc = u @ W_x ; split -> delta-input (bf16), B|C (fp32)
  gemm_bf16_kernel<EPI_DBC>
      <<<dim3((DTR_ + 2 * NS_ + BN - 1) / BN, M_ / BM), blk, 0, stream>>>(
          u_bf, WxT_bf, DTR_ + 2 * NS_, DIN_, DIN_, nullptr, bc_f, nullptr, ddt_bf);

  // 5) delta = softplus(ddt @ W_dt + b_dt)
  gemm_bf16_kernel<EPI_SOFTPLUS>
      <<<dim3(DIN_ / BN, M_ / BM), blk, 0, stream>>>(
          ddt_bf, WdtT_bf, DIN_, DTR_, DTR_, b_dt, delta_f, nullptr, nullptr);

  // 6) selective scan + gating -> g (bf16 feed for final GEMM)
  scan_kernel<<<(B_ * DIN_ + 255) / 256, blk, 0, stream>>>(
      delta_f, u_f, bc_f, r_f, A_log, Dp, g_bf);

  // 7) out = g @ W_out + xs
  gemm_bf16_kernel<EPI_ADD>
      <<<dim3(DM_ / BN, M_ / BM), blk, 0, stream>>>(
          g_bf, WoutT_bf, DM_, DIN_, DIN_, xs_f, out, nullptr, nullptr);
}
